// SingleHeadAttention_27393301414385
// MI455X (gfx1250) — matrix operand order repaired and checked by reading, not compiled, not run on hardware
//
#include <hip/hip_runtime.h>

// MI455X (gfx1250) single-head attention with group-weight modulation.
// Compute-bound (~264 GFLOP vs ~0.5 GB of HBM traffic) -> everything big runs
// on V_WMMA_F32_16X16X32_BF16 (bf16 in, f32 accumulate). The G-normalization
// constants cancel against the final renormalization, so the attention kernel
// is a flash-style single pass: P = exp(s*scale - m) * G, running (m, l),
// out = (P @ v) / l.
//
// All global->LDS tile staging uses GLOBAL_LOAD_ASYNC_TO_LDS_B128 (ASYNCcnt
// path, no VGPR roundtrip) and is double-buffered so the DMA overlaps WMMA.
// v is pre-transposed once into vT[b][d][n] so the P@v B-tiles are also a
// straight async copy.

#define BATCH 32
#define SEQ   1024
#define DIMD  768
#define GPDIM 49
#define QKVN  (3 * DIMD)       // 2304
#define ROWS  (BATCH * SEQ)    // 32768

typedef __bf16 bf16_t;
typedef __attribute__((ext_vector_type(16))) __bf16 v16bf;
typedef __attribute__((ext_vector_type(8)))  __bf16 v8bf;
typedef __attribute__((ext_vector_type(8)))  float  v8f;
typedef __attribute__((ext_vector_type(4)))  float  v4f;

static __device__ __forceinline__ v8f wmma_bf16(v16bf a, v16bf b, v8f c) {
  // D = A(16x32 bf16) * B(32x16 bf16) + C(16x16 f32)
  return __builtin_amdgcn_wmma_f32_16x16x32_bf16(false, a, false, b,
                                                 (short)0, c, false, false);
}

// Direct global->LDS async copy of 16 bytes per lane (CDNA5 ASYNCcnt path).
static __device__ __forceinline__ void async_copy_b128(bf16_t* ldst,
                                                       const bf16_t* gsrc) {
  unsigned int       l = (unsigned int)(unsigned long long)ldst;
  unsigned long long g = (unsigned long long)gsrc;
  asm volatile("global_load_async_to_lds_b128 %0, %1, off"
               :: "v"(l), "v"(g) : "memory");
}
static __device__ __forceinline__ void wait_async0() {
  asm volatile("s_wait_asynccnt 0" ::: "memory");
}

// A-fragment (16x32 bf16) from a row-major LDS tile (row stride `ld` elems).
// ISA layout: lane m = l&15, half h = l>>4; VGPR0-3: K = h*8+0..7,
// VGPR4-7: K = 16+h*8+0..7  -> two contiguous 16B reads per lane.
static __device__ __forceinline__ v16bf load_a_frag(const bf16_t* tile, int ld,
                                                    int row0, int k0) {
  const int lane = threadIdx.x & 31;
  const int m = lane & 15, h = lane >> 4;
  const bf16_t* p = tile + (long)(row0 + m) * ld + k0 + h * 8;
  v8bf lo = *(const v8bf*)(p);
  v8bf hi = *(const v8bf*)(p + 16);
  v16bf f;
#pragma unroll
  for (int i = 0; i < 8; ++i) { f[i] = lo[i]; f[i + 8] = hi[i]; }
  return f;
}

// B-fragment (32x16 bf16) from an N-major LDS tile tile[n][k] (stride ldk).
// ISA layout: lane n = l&15, half h = l>>4; K = h*8+0..7, 16+h*8+0..7.
static __device__ __forceinline__ v16bf load_b_frag(const bf16_t* tileN, int ldk,
                                                    int n0, int k0) {
  const int lane = threadIdx.x & 31;
  const int n = lane & 15, h = lane >> 4;
  const bf16_t* p = tileN + (long)(n0 + n) * ldk + k0 + h * 8;
  v8bf lo = *(const v8bf*)(p);
  v8bf hi = *(const v8bf*)(p + 16);
  v16bf f;
#pragma unroll
  for (int i = 0; i < 8; ++i) { f[i] = lo[i]; f[i + 8] = hi[i]; }
  return f;
}

// ---------------------------------------------------------------------------
// Kernel 1: qkv = bf16(x @ w_qkv + b_qkv).  Block tile 128(M) x 64(N), K=768.
// 8 waves as 4(M)x2(N); each wave owns a 32x32 tile (4 accumulators).
// Both stages convert f32->bf16 in flight, so they keep the VGPR path.
// ---------------------------------------------------------------------------
__global__ __launch_bounds__(256, 2)
void qkv_gemm(const float* __restrict__ x, const float* __restrict__ wq,
              const float* __restrict__ bias, bf16_t* __restrict__ out) {
  extern __shared__ char smem[];
  bf16_t* At = (bf16_t*)smem;            // [128][32] row-major
  bf16_t* Bt = (bf16_t*)(smem + 8192);   // [64][32]  n-major (transposed stage)
  const int tid = threadIdx.x, lane = tid & 31, wv = tid >> 5;
  const int wm = wv >> 1, wn = wv & 1;
  const int hf = lane >> 4, nl = lane & 15;
  const long m0 = (long)blockIdx.y * 128;
  const int  n0 = blockIdx.x * 64;

  v8f acc[2][2];
#pragma unroll
  for (int i = 0; i < 2; ++i)
#pragma unroll
    for (int j = 0; j < 2; ++j)
#pragma unroll
      for (int r = 0; r < 8; ++r) acc[i][j][r] = 0.0f;

  for (int k0 = 0; k0 < DIMD; k0 += 32) {
    __syncthreads();
    for (int i = tid; i < 128 * 8; i += 256) {      // A: f32 -> bf16
      int r = i >> 3, c = (i & 7) * 4;
      v4f v = *(const v4f*)&x[(m0 + r) * DIMD + k0 + c];
#pragma unroll
      for (int e = 0; e < 4; ++e) At[r * 32 + c + e] = (bf16_t)v[e];
    }
    for (int i = tid; i < 32 * 16; i += 256) {      // B: f32 -> bf16, transpose
      int r = i >> 4, c = (i & 15) * 4;             // k row r, n col c
      v4f v = *(const v4f*)&wq[(long)(k0 + r) * QKVN + n0 + c];
#pragma unroll
      for (int e = 0; e < 4; ++e) Bt[(c + e) * 32 + r] = (bf16_t)v[e];
    }
    __syncthreads();
#pragma unroll
    for (int i = 0; i < 2; ++i) {
      v16bf a = load_a_frag(At, 32, wm * 32 + i * 16, 0);
#pragma unroll
      for (int j = 0; j < 2; ++j) {
        v16bf b = load_b_frag(Bt, 32, wn * 32 + j * 16, 0);
        acc[i][j] = wmma_bf16(a, b, acc[i][j]);
      }
    }
  }
#pragma unroll
  for (int i = 0; i < 2; ++i)
#pragma unroll
    for (int j = 0; j < 2; ++j)
#pragma unroll
      for (int r = 0; r < 8; ++r) {
        int row = wm * 32 + i * 16 + hf * 8 + r;
        int col = wn * 32 + j * 16 + nl;
        out[(m0 + row) * QKVN + n0 + col] = (bf16_t)(acc[i][j][r] + bias[n0 + col]);
      }
}

// ---------------------------------------------------------------------------
// Kernel 2: gw = softmax(v @ w_gp) padded with zeros to 64 columns (bf16).
// Tiny (~2.5 GFLOP) -> plain VALU. One 64-thread block per row.
// ---------------------------------------------------------------------------
__global__ __launch_bounds__(64)
void gw_kernel(const bf16_t* __restrict__ qkv, const float* __restrict__ wgp,
               bf16_t* __restrict__ gw) {
  __shared__ float  sbuf[64];
  __shared__ bf16_t vrow[DIMD];
  const long row = blockIdx.x;
  const int  t   = threadIdx.x;
  const bf16_t* vsrc = &qkv[row * QKVN + 2 * DIMD];
  for (int i = t; i < DIMD / 8; i += 64)
    async_copy_b128(&vrow[i * 8], &vsrc[i * 8]);
  wait_async0();
  __syncthreads();

  float acc = -3.0e38f;
  if (t < GPDIM) {
    acc = 0.0f;
    for (int d = 0; d < DIMD; ++d) acc += (float)vrow[d] * wgp[d * GPDIM + t];
  }
  sbuf[t] = acc;
  __syncthreads();
  for (int off = 32; off > 0; off >>= 1) {
    if (t < off) sbuf[t] = fmaxf(sbuf[t], sbuf[t + off]);
    __syncthreads();
  }
  float mx = sbuf[0];
  __syncthreads();
  float e = (t < GPDIM) ? __expf(acc - mx) : 0.0f;
  sbuf[t] = e;
  __syncthreads();
  for (int off = 32; off > 0; off >>= 1) {
    if (t < off) sbuf[t] += sbuf[t + off];
    __syncthreads();
  }
  float inv = 1.0f / (sbuf[0] + 1e-20f);
  gw[row * 64 + t] = (bf16_t)(e * inv);
}

// ---------------------------------------------------------------------------
// Kernel 2b: vT[b][d][n] = v[b][n][d]  (one-time 64x64-tile transpose through
// padded LDS, coalesced on both sides). Lets the attention P@v B-tiles be a
// straight async DMA instead of a per-tile VGPR transpose.
// ---------------------------------------------------------------------------
__global__ __launch_bounds__(256)
void vtrans_kernel(const bf16_t* __restrict__ qkv, bf16_t* __restrict__ vt) {
  __shared__ bf16_t t[64][72];            // pad 64->72 to dodge bank conflicts
  const int b  = blockIdx.z;
  const int d0 = blockIdx.x * 64, n0 = blockIdx.y * 64;
  for (int i = threadIdx.x; i < 64 * 8; i += 256) {
    int r = i >> 3, c = (i & 7) * 8;      // v row n0+r, d cols d0+c..+7
    v8bf v = *(const v8bf*)&qkv[((long)b * SEQ + n0 + r) * QKVN + 2 * DIMD + d0 + c];
#pragma unroll
    for (int e = 0; e < 8; ++e) t[r][c + e] = v[e];
  }
  __syncthreads();
  for (int i = threadIdx.x; i < 64 * 8; i += 256) {
    int r = i >> 3, c = (i & 7) * 8;      // vt row d0+r, n cols n0+c..+7
    v8bf o;
#pragma unroll
    for (int e = 0; e < 8; ++e) o[e] = t[c + e][r];
    *(v8bf*)&vt[((long)b * DIMD + d0 + r) * SEQ + n0 + c] = o;
  }
}

// ---------------------------------------------------------------------------
// Kernel 3: flash-style attention with G modulation.
// Block = (batch, 32-row q tile), 8 waves as 2(M=16 rows)x4(N groups).
// q tile LDS-resident (48KB); k and vT streamed in 128-wide chunks through
// two ping-pong 16KB buffers: issue chunk dc+1 right after the barrier that
// frees its buffer, compute on chunk dc -> DMA overlaps WMMA.
// ---------------------------------------------------------------------------
#define ATTN_SMEM 99200

__global__ __launch_bounds__(256, 1)
void attn_kernel(const bf16_t* __restrict__ qkv, const bf16_t* __restrict__ gw,
                 const bf16_t* __restrict__ vt, bf16_t* __restrict__ attnOut) {
  extern __shared__ char smem[];
  bf16_t* qT   = (bf16_t*)(smem);            // [32][768]
  bf16_t* bufA = (bf16_t*)(smem + 49152);    // [64][128] (k) / [128][64] (vT)
  bf16_t* bufB = (bf16_t*)(smem + 65536);    // ping-pong partner
  bf16_t* gwK  = (bf16_t*)(smem + 81920);    // [64][64]
  bf16_t* gwQ  = (bf16_t*)(smem + 90112);    // [32][64]
  bf16_t* Pb   = (bf16_t*)(smem + 94208);    // [32][64] bf16 P tile
  float*  mrow = (float*)(smem + 98304);     // [32] running max
  float*  lrow = mrow + 32;                  // [32] running denom
  float*  facr = lrow + 32;                  // [32] rescale factor
  float*  cmax = facr + 32;                  // [4][32] per-wave col-group max
  bf16_t* bufs[2] = { bufA, bufB };

  const int tid  = threadIdx.x;
  const int lane = tid & 31;
  const int wv   = tid >> 5;
  const int wm   = wv & 1;        // q-row group: rows wm*16 .. +16
  const int wn   = wv >> 1;       // column group (0..3)
  const int hf   = lane >> 4;
  const int nl   = lane & 15;
  const int b    = blockIdx.y;
  const long rowQ = (long)b * SEQ + (long)blockIdx.x * 32;

  for (int i = tid; i < 32 * 96; i += 256) {                 // stage q tile
    int r = i / 96, c = (i % 96) * 8;
    async_copy_b128(&qT[r * DIMD + c], &qkv[(rowQ + r) * QKVN + c]);
  }
  for (int i = tid; i < 32 * 8; i += 256) {                  // stage gw(q)
    int r = i >> 3, c = (i & 7) * 8;
    async_copy_b128(&gwQ[r * 64 + c], &gw[(rowQ + r) * 64 + c]);
  }
  if (tid < 32) { mrow[tid] = -3.0e38f; lrow[tid] = 0.0f; }

  v8f outAcc[6][2];
#pragma unroll
  for (int ch = 0; ch < 6; ++ch)
#pragma unroll
    for (int j = 0; j < 2; ++j)
#pragma unroll
      for (int r = 0; r < 8; ++r) outAcc[ch][j][r] = 0.0f;

  const float scale = rsqrtf((float)DIMD);

  for (int kt = 0; kt < SEQ / 64; ++kt) {
    const long rowK = (long)b * SEQ + (long)kt * 64;

    // Close out the previous iteration (all buffers free), then prefetch
    // gw(k) and S-chunk 0.  (For kt==0 this also covers the prologue.)
    wait_async0();
    __syncthreads();
    for (int i = tid; i < 64 * 8; i += 256) {                // stage gw(k)
      int r = i >> 3, c = (i & 7) * 8;
      async_copy_b128(&gwK[r * 64 + c], &gw[(rowK + r) * 64 + c]);
    }
    for (int i = tid; i < 64 * 16; i += 256) {               // S chunk 0
      int r = i >> 4, c = (i & 15) * 8;
      async_copy_b128(&bufA[r * 128 + c], &qkv[(rowK + r) * QKVN + DIMD + c]);
    }

    v8f gAcc, sAcc;
#pragma unroll
    for (int r = 0; r < 8; ++r) { gAcc[r] = 0.0f; sAcc[r] = 0.0f; }

    // S tile = q @ k^T over DIM in 128-wide double-buffered chunks
    for (int dc = 0; dc < 6; ++dc) {
      wait_async0();              // chunk dc (and at dc==0 also gwK) landed
      __syncthreads();            // ...in every wave; buffer dc+1 is free
      if (dc + 1 < 6) {
        bf16_t* nb = bufs[(dc + 1) & 1];
        for (int i = tid; i < 64 * 16; i += 256) {
          int r = i >> 4, c = (i & 15) * 8;
          async_copy_b128(&nb[r * 128 + c],
                          &qkv[(rowK + r) * QKVN + DIMD + (dc + 1) * 128 + c]);
        }
      }
      if (dc == 0) {
        // G tile = gwQ @ gwK^T  (K = 64, zero-padded 49..63)
#pragma unroll
        for (int s = 0; s < 2; ++s) {
          v16bf a = load_a_frag(gwQ, 64, wm * 16, s * 32);
          v16bf bb = load_b_frag(gwK, 64, wn * 16, s * 32);
          gAcc = wmma_bf16(a, bb, gAcc);
        }
      }
      bf16_t* cb = bufs[dc & 1];
#pragma unroll
      for (int s = 0; s < 4; ++s) {
        v16bf a = load_a_frag(qT, DIMD, wm * 16, dc * 128 + s * 32);
        v16bf bb = load_b_frag(cb, 128, wn * 16, s * 32);
        sAcc = wmma_bf16(a, bb, sAcc);
      }
    }

    // Prefetch PV chunk 0 into bufA (freed by the dc==5 entry barrier);
    // the DMA overlaps the whole softmax epilogue below.
    for (int i = tid; i < 128 * 8; i += 256) {
      int r = i >> 3, c = (i & 7) * 8;    // vt row d=r, n cols c..c+7
      async_copy_b128(&bufA[r * 64 + c],
                      &vt[((long)b * DIMD + r) * SEQ + kt * 64 + c]);
    }

    // --- online-softmax epilogue: tile row max (16-lane + cross-wave) ---
    float sv[8];
#pragma unroll
    for (int r = 0; r < 8; ++r) {
      float v = sAcc[r] * scale;
      sv[r] = v;
      float m = v;
#pragma unroll
      for (int off = 8; off > 0; off >>= 1) m = fmaxf(m, __shfl_xor(m, off, 32));
      if (nl == 0) cmax[wn * 32 + wm * 16 + hf * 8 + r] = m;
    }
    __syncthreads();
    if (tid < 32) {
      float t = fmaxf(fmaxf(cmax[tid], cmax[32 + tid]),
                      fmaxf(cmax[64 + tid], cmax[96 + tid]));
      float mo = mrow[tid];
      float mn = fmaxf(mo, t);
      float f  = __expf(mo - mn);
      mrow[tid] = mn;
      lrow[tid] *= f;
      facr[tid] = f;
    }
    __syncthreads();
    {
      float f8[8], mn8[8];
#pragma unroll
      for (int r = 0; r < 8; ++r) {
        int row = wm * 16 + hf * 8 + r;
        f8[r]  = facr[row];
        mn8[r] = mrow[row];
      }
#pragma unroll
      for (int r = 0; r < 8; ++r) {
        int row = wm * 16 + hf * 8 + r;
        float p = __expf(sv[r] - mn8[r]) * gAcc[r];     // G >= 0
        Pb[row * 64 + wn * 16 + nl] = (bf16_t)p;
        float s = p;
#pragma unroll
        for (int off = 8; off > 0; off >>= 1) s += __shfl_xor(s, off, 32);
        if (nl == 0) atomicAdd(&lrow[row], s);
      }
#pragma unroll
      for (int ch = 0; ch < 6; ++ch)
#pragma unroll
        for (int j = 0; j < 2; ++j)
#pragma unroll
          for (int r = 0; r < 8; ++r) outAcc[ch][j][r] *= f8[r];
    }

    // --- out += P @ v : double-buffered async vT chunks ---
#pragma unroll
    for (int ch = 0; ch < 6; ++ch) {
      wait_async0();              // PV chunk ch landed
      __syncthreads();            // Pb visible (ch==0) / buffer ch+1 free
      if (ch + 1 < 6) {
        bf16_t* nb = bufs[(ch + 1) & 1];
        for (int i = tid; i < 128 * 8; i += 256) {
          int r = i >> 3, c = (i & 7) * 8;
          async_copy_b128(&nb[r * 64 + c],
                          &vt[((long)b * DIMD + (ch + 1) * 128 + r) * SEQ + kt * 64 + c]);
        }
      }
      bf16_t* cb = bufs[ch & 1];
#pragma unroll
      for (int s = 0; s < 2; ++s) {
        v16bf a = load_a_frag(Pb, 64, wm * 16, s * 32);
#pragma unroll
        for (int j = 0; j < 2; ++j) {
          v16bf bb = load_b_frag(cb, 64, wn * 32 + j * 16, s * 32);
          outAcc[ch][j] = wmma_bf16(a, bb, outAcc[ch][j]);
        }
      }
    }
  }

  __syncthreads();
  {
    float inv8[8];
#pragma unroll
    for (int r = 0; r < 8; ++r)
      inv8[r] = 1.0f / (lrow[wm * 16 + hf * 8 + r] + 1e-8f);
#pragma unroll
    for (int ch = 0; ch < 6; ++ch)
#pragma unroll
      for (int j = 0; j < 2; ++j)
#pragma unroll
        for (int r = 0; r < 8; ++r) {
          int row = wm * 16 + hf * 8 + r;
          int col = ch * 128 + wn * 32 + j * 16 + nl;
          attnOut[(rowQ + row) * DIMD + col] = (bf16_t)(outAcc[ch][j][r] * inv8[r]);
        }
  }
}

// ---------------------------------------------------------------------------
// Kernel 4: out = attnOut(bf16) @ w_proj + b_proj, f32 result.
// A tile is a pure bf16 copy -> async DMA; B tile converts f32->bf16.
// ---------------------------------------------------------------------------
__global__ __launch_bounds__(256, 2)
void proj_gemm(const bf16_t* __restrict__ a, const float* __restrict__ wp,
               const float* __restrict__ bias, float* __restrict__ out) {
  extern __shared__ char smem[];
  bf16_t* At = (bf16_t*)smem;            // [128][32]
  bf16_t* Bt = (bf16_t*)(smem + 8192);   // [64][32] n-major
  const int tid = threadIdx.x, lane = tid & 31, wv = tid >> 5;
  const int wm = wv >> 1, wn = wv & 1;
  const int hf = lane >> 4, nl = lane & 15;
  const long m0 = (long)blockIdx.y * 128;
  const int  n0 = blockIdx.x * 64;

  v8f acc[2][2];
#pragma unroll
  for (int i = 0; i < 2; ++i)
#pragma unroll
    for (int j = 0; j < 2; ++j)
#pragma unroll
      for (int r = 0; r < 8; ++r) acc[i][j][r] = 0.0f;

  for (int k0 = 0; k0 < DIMD; k0 += 32) {
    __syncthreads();
    for (int i = tid; i < 128 * 4; i += 256) {       // A (already bf16): async
      int r = i >> 2, c = (i & 3) * 8;
      async_copy_b128(&At[r * 32 + c], &a[(m0 + r) * DIMD + k0 + c]);
    }
    for (int i = tid; i < 32 * 16; i += 256) {       // B: f32 -> bf16, transpose
      int r = i >> 4, c = (i & 15) * 4;
      v4f v = *(const v4f*)&wp[(long)(k0 + r) * DIMD + n0 + c];
#pragma unroll
      for (int e = 0; e < 4; ++e) Bt[(c + e) * 32 + r] = (bf16_t)v[e];
    }
    wait_async0();
    __syncthreads();
#pragma unroll
    for (int i = 0; i < 2; ++i) {
      v16bf af = load_a_frag(At, 32, wm * 32 + i * 16, 0);
#pragma unroll
      for (int j = 0; j < 2; ++j) {
        v16bf bf = load_b_frag(Bt, 32, wn * 32 + j * 16, 0);
        acc[i][j] = wmma_bf16(af, bf, acc[i][j]);
      }
    }
  }
#pragma unroll
  for (int i = 0; i < 2; ++i)
#pragma unroll
    for (int j = 0; j < 2; ++j)
#pragma unroll
      for (int r = 0; r < 8; ++r) {
        int row = wm * 32 + i * 16 + hf * 8 + r;
        int col = wn * 32 + j * 16 + nl;
        out[(m0 + row) * DIMD + n0 + col] = acc[i][j][r] + bias[n0 + col];
      }
}

// ---------------------------------------------------------------------------
extern "C" void kernel_launch(void* const* d_in, const int* in_sizes, int n_in,
                              void* d_out, int out_size, void* d_ws, size_t ws_size,
                              hipStream_t stream) {
  (void)in_sizes; (void)n_in; (void)out_size; (void)ws_size;
  const float* x      = (const float*)d_in[0];
  const float* w_qkv  = (const float*)d_in[1];
  const float* b_qkv  = (const float*)d_in[2];
  const float* w_proj = (const float*)d_in[3];
  const float* b_proj = (const float*)d_in[4];
  const float* w_gp   = (const float*)d_in[5];
  float* out = (float*)d_out;

  // Workspace (bf16): qkv [32768][2304] 144MB, gw [32768][64] 4MB,
  // attnOut [32768][768] 48MB, vT [32][768][1024] 48MB. Total ~244MB.
  char* p = (char*)d_ws;
  bf16_t* qkv = (bf16_t*)p; p += (size_t)ROWS * QKVN * sizeof(bf16_t);
  bf16_t* gw  = (bf16_t*)p; p += (size_t)ROWS * 64 * sizeof(bf16_t);
  bf16_t* ao  = (bf16_t*)p; p += (size_t)ROWS * DIMD * sizeof(bf16_t);
  bf16_t* vt  = (bf16_t*)p;

  hipFuncSetAttribute((const void*)attn_kernel,
                      hipFuncAttributeMaxDynamicSharedMemorySize, ATTN_SMEM);

  qkv_gemm<<<dim3(QKVN / 64, ROWS / 128), 256, 12 * 1024, stream>>>(x, w_qkv, b_qkv, qkv);
  gw_kernel<<<dim3(ROWS), 64, 0, stream>>>(qkv, w_gp, gw);
  vtrans_kernel<<<dim3(DIMD / 64, SEQ / 64, BATCH), 256, 0, stream>>>(qkv, vt);
  attn_kernel<<<dim3(SEQ / 32, BATCH), 256, ATTN_SMEM, stream>>>(qkv, gw, vt, ao);
  proj_gemm<<<dim3(DIMD / 64, ROWS / 128), 256, 12 * 1024, stream>>>(ao, w_proj, b_proj, out);
}
